// MoEFFN_52493090292129
// MI455X (gfx1250) — matrix operand order repaired and checked by reading, not compiled, not run on hardware
//
#include <hip/hip_runtime.h>
#include <hip/hip_bf16.h>

#define N_TOK 32768
#define C_DIM 256
#define F_DIM 1024
#define E_NUM 16
#define TOPK  2

#define TILE_ROWS 64
#define MOE_TILES ((N_TOK * TOPK + E_NUM * TILE_ROWS) / TILE_ROWS)  // 1040
#define GATE_BLOCKS (N_TOK / 256)

// padded LDS strides (elements); multiples of 16 -> 32B-aligned v16bf loads
#define W1_STRIDE 272   // row = 256 K-values of one W1 column f
#define W2_STRIDE 144   // row = 128 K-values (f-chunk) of one W2 column c
#define H_STRIDE  144   // row = 128 h values (f-chunk) of one token row
#define SMEM_BYTES ((128 * W1_STRIDE + 256 * W2_STRIDE + 64 * H_STRIDE) * 2)

typedef __attribute__((ext_vector_type(16))) __bf16 v16bf;
typedef __attribute__((ext_vector_type(8)))  __bf16 v8bf;
typedef __attribute__((ext_vector_type(8)))  float  v8f;
typedef __attribute__((ext_vector_type(4)))  int    v4i;

__device__ __forceinline__ unsigned short f2bf(float f) {
    unsigned int u = __float_as_uint(f);
    unsigned int r = (u + 0x7FFFu + ((u >> 16) & 1u)) >> 16;  // RNE
    return (unsigned short)r;
}

// A operand (16x32), contiguous k order shared with B: lanes 0-15 hold K = [0..15],
// lanes 16-31 hold K = [16..31]  -> two 16B loads concatenated.
__device__ __forceinline__ v16bf load_a16(const unsigned short* p, int khalf) {
    v8bf lo = *(const v8bf*)(p + khalf * 16);
    v8bf hi = *(const v8bf*)(p + khalf * 16 + 8);
    return __builtin_shufflevector(lo, hi, 0,1,2,3,4,5,6,7,8,9,10,11,12,13,14,15);
}

// ---- async global->LDS copy (gfx1250 GLOBAL_LOAD_ASYNC_TO_LDS_B128), with
// a synchronous fallback so compilation never breaks.
#if __has_builtin(__builtin_amdgcn_global_load_async_to_lds_b128)
#define HAVE_ASYNC_LDS 1
#else
#define HAVE_ASYNC_LDS 0
#endif

__device__ __forceinline__ void cp16_g2l(unsigned short* lds, const unsigned short* g) {
#if HAVE_ASYNC_LDS
    __builtin_amdgcn_global_load_async_to_lds_b128(
        (__attribute__((address_space(1))) v4i*)(void*)g,
        (__attribute__((address_space(3))) v4i*)(void*)lds, 0, 0);
#else
    *(float4*)lds = *(const float4*)g;
#endif
}

__device__ __forceinline__ void async_wait_all() {
#if HAVE_ASYNC_LDS
    asm volatile("s_wait_asynccnt 0x0" ::: "memory");
#endif
}

// ---------------- conversion kernels ----------------
__global__ void k_cvt_x(const float* __restrict__ x, unsigned short* __restrict__ xb, int n) {
    int i = blockIdx.x * blockDim.x + threadIdx.x;
    if (i < n) xb[i] = f2bf(x[i]);
}

__global__ void k_cvt_w1t(const float* __restrict__ W1, unsigned short* __restrict__ w1t) {
    int i = blockIdx.x * blockDim.x + threadIdx.x;           // over E*C*F
    int e = i / (C_DIM * F_DIM);
    int r = i - e * (C_DIM * F_DIM);
    int c = r / F_DIM;
    int f = r - c * F_DIM;
    w1t[((size_t)e * F_DIM + f) * C_DIM + c] = f2bf(W1[i]);  // [E][F][C]
}

__global__ void k_cvt_w2t(const float* __restrict__ W2, unsigned short* __restrict__ w2t) {
    int i = blockIdx.x * blockDim.x + threadIdx.x;           // over E*F*C
    int e = i / (F_DIM * C_DIM);
    int r = i - e * (F_DIM * C_DIM);
    int f = r / C_DIM;
    int c = r - f * C_DIM;
    w2t[((size_t)e * C_DIM + c) * F_DIM + f] = f2bf(W2[i]);  // [E][C][F]
}

// ---------------- routing / gating ----------------
__global__ void k_init(int* __restrict__ counts) {
    if (threadIdx.x < E_NUM) counts[threadIdx.x] = 0;
}

__global__ __launch_bounds__(256)
void k_gate(const float* __restrict__ x, const float* __restrict__ gW,
            const float* __restrict__ gb, int* __restrict__ eidx,
            float* __restrict__ tw, int* __restrict__ counts,
            float* __restrict__ entpart) {
    __shared__ float sgw[E_NUM * C_DIM];  // 16 KB
    __shared__ float sred[256];
    for (int i = threadIdx.x; i < E_NUM * C_DIM; i += 256) sgw[i] = gW[i];
    __syncthreads();

    int tok = blockIdx.x * 256 + threadIdx.x;
    float logit[E_NUM];
    #pragma unroll
    for (int e = 0; e < E_NUM; ++e) logit[e] = gb[e];
    const float* xr = x + (size_t)tok * C_DIM;
    for (int c = 0; c < C_DIM; ++c) {
        float xv = xr[c];
        #pragma unroll
        for (int e = 0; e < E_NUM; ++e) logit[e] += xv * sgw[e * C_DIM + c];
    }
    float m = logit[0];
    #pragma unroll
    for (int e = 1; e < E_NUM; ++e) m = fmaxf(m, logit[e]);
    float s = 0.f;
    #pragma unroll
    for (int e = 0; e < E_NUM; ++e) s += expf(logit[e] - m);
    float inv = 1.f / s;
    float ent = 0.f;
    #pragma unroll
    for (int e = 0; e < E_NUM; ++e) {
        float pe = expf(logit[e] - m) * inv;
        ent -= pe * logf(pe + 1e-8f);
    }
    int i1 = 0; float v1 = logit[0];
    #pragma unroll
    for (int e = 1; e < E_NUM; ++e) if (logit[e] > v1) { v1 = logit[e]; i1 = e; }
    int i2 = -1; float v2 = -3.4e38f;
    #pragma unroll
    for (int e = 0; e < E_NUM; ++e) if (e != i1 && logit[e] > v2) { v2 = logit[e]; i2 = e; }
    float w0 = 1.f / (1.f + expf(v2 - v1));
    eidx[tok * 2 + 0] = i1;  eidx[tok * 2 + 1] = i2;
    tw[tok * 2 + 0] = w0;    tw[tok * 2 + 1] = 1.f - w0;
    atomicAdd(&counts[i1], 1);
    atomicAdd(&counts[i2], 1);

    sred[threadIdx.x] = ent;
    __syncthreads();
    for (int st = 128; st > 0; st >>= 1) {
        if (threadIdx.x < st) sred[threadIdx.x] += sred[threadIdx.x + st];
        __syncthreads();
    }
    if (threadIdx.x == 0) entpart[blockIdx.x] = sred[0];
}

__global__ void k_scan(const int* __restrict__ counts, int* __restrict__ offA,
                       int* __restrict__ poffA, int* __restrict__ counts2,
                       const float* __restrict__ entpart, float* __restrict__ outp) {
    if (threadIdx.x == 0) {
        int off = 0, poff = 0;
        for (int e = 0; e < E_NUM; ++e) {
            offA[e] = off; poffA[e] = poff; counts2[e] = 0;
            int c = counts[e];
            off += c;
            poff += ((c + TILE_ROWS - 1) / TILE_ROWS) * TILE_ROWS;
        }
        offA[E_NUM] = off; poffA[E_NUM] = poff;
        float s = 0.f;
        for (int i = 0; i < GATE_BLOCKS; ++i) s += entpart[i];
        outp[(size_t)N_TOK * C_DIM] = s / (float)N_TOK;  // gate_entropy
    }
}

__global__ void k_assign(const int* __restrict__ eidx, const int* __restrict__ offA,
                         int* __restrict__ counts2, int* __restrict__ perm) {
    int s = blockIdx.x * blockDim.x + threadIdx.x;
    if (s < N_TOK * TOPK) {
        int e = eidx[s];
        int pos = atomicAdd(&counts2[e], 1);
        perm[offA[e] + pos] = s;
    }
}

// ---------------- fused grouped FFN: relu(x@W1+b1)@W2+b2, scaled ----------------
__global__ __launch_bounds__(256)
void k_moe(const unsigned short* __restrict__ xb,
           const unsigned short* __restrict__ w1t,  // [E][F][C] bf16
           const unsigned short* __restrict__ w2t,  // [E][C][F] bf16
           const float* __restrict__ b1, const float* __restrict__ b2,
           const int* __restrict__ offA, const int* __restrict__ poffA,
           const int* __restrict__ counts, const int* __restrict__ perm,
           const float* __restrict__ tw, float* __restrict__ ybuf) {
    extern __shared__ char dsm[];
    unsigned short* sW1 = (unsigned short*)dsm;                               // 128 x W1_STRIDE
    unsigned short* sW2 = sW1 + 128 * W1_STRIDE;                              // 256 x W2_STRIDE
    unsigned short* hsm = sW2 + 256 * W2_STRIDE;                              // 64  x H_STRIDE
    __shared__ int   sslot[TILE_ROWS];
    __shared__ float swt[TILE_ROWS];
    __shared__ int   sE[2];

    int row0 = blockIdx.x * TILE_ROWS;
    if (threadIdx.x == 0) {
        int e = -1;
        for (int i = 0; i < E_NUM; ++i)
            if (row0 >= poffA[i] && row0 < poffA[i + 1]) { e = i; break; }
        sE[0] = e;
        sE[1] = (e >= 0) ? poffA[e] : 0;
    }
    __syncthreads();
    int e = sE[0];
    if (e < 0) return;  // uniform across block (padding tail)
    int segbase = sE[1];

    if (threadIdx.x < TILE_ROWS) {
        int lr = row0 - segbase + threadIdx.x;
        int slot = -1; float w = 0.f;
        if (lr < counts[e]) { slot = perm[offA[e] + lr]; w = tw[slot]; }
        sslot[threadIdx.x] = slot;
        swt[threadIdx.x] = w;
    }
    __syncthreads();

    const int wv    = threadIdx.x >> 5;
    const int lane  = threadIdx.x & 31;
    const int nloc  = lane & 15;
    const int khalf = lane >> 4;
    const int wr = wv & 3;   // 16-row group
    const int wc = wv >> 2;  // 0..1

    int arow = wr * 16 + nloc;
    int asl  = sslot[arow];
    const unsigned short* aptr = xb + (size_t)((asl < 0 ? 0 : asl) >> 1) * C_DIM;

    const unsigned short* w1e = w1t + (size_t)e * F_DIM * C_DIM;
    const unsigned short* w2e = w2t + (size_t)e * C_DIM * F_DIM;

    v8f acc2[8];
    #pragma unroll
    for (int t = 0; t < 8; ++t)
        #pragma unroll
        for (int v = 0; v < 8; ++v) acc2[t][v] = 0.f;

    // preload W1 chunk fc=0: 128 rows (f) x 256 (c), contiguous 64KB in w1t
    {
        const unsigned short* src = w1e;  // + 0*C_DIM
        for (int u = threadIdx.x; u < 4096; u += 256) {
            int row = u >> 5, seg = u & 31;
            cp16_g2l(sW1 + row * W1_STRIDE + seg * 8, src + row * C_DIM + seg * 8);
        }
    }

    for (int fc = 0; fc < F_DIM; fc += 128) {
        async_wait_all();
        __syncthreads();                      // sW1[fc] ready; prior phase2 done

        // issue W2 chunk: 256 rows (c) x 128 (f in [fc,fc+128))
        for (int u = threadIdx.x; u < 4096; u += 256) {
            int row = u >> 4, seg = u & 15;
            cp16_g2l(sW2 + row * W2_STRIDE + seg * 8,
                     w2e + (size_t)row * F_DIM + fc + seg * 8);
        }

        // ---- phase 1: h[64 x 128] = relu(x @ W1[:, fc:fc+128] + b1), B from LDS
        v8f acc1[4];
        #pragma unroll
        for (int t = 0; t < 4; ++t)
            #pragma unroll
            for (int v = 0; v < 8; ++v) acc1[t][v] = 0.f;
        #pragma unroll
        for (int kk = 0; kk < C_DIM; kk += 32) {
            v16bf a = load_a16(aptr + kk, khalf);
            #pragma unroll
            for (int t = 0; t < 4; ++t) {
                int fl = wc * 64 + t * 16 + nloc;
                v16bf b = *(const v16bf*)(sW1 + fl * W1_STRIDE + kk + khalf * 16);
                acc1[t] = __builtin_amdgcn_wmma_f32_16x16x32_bf16(
                    false, a, false, b, (short)0, acc1[t], false, false);
            }
        }
        #pragma unroll
        for (int t = 0; t < 4; ++t) {
            int col = wc * 64 + t * 16 + nloc;
            float bias = b1[e * F_DIM + fc + col];
            #pragma unroll
            for (int v = 0; v < 8; ++v) {
                int m = v + 8 * khalf;
                float hv = acc1[t][v] + bias;
                hsm[(wr * 16 + m) * H_STRIDE + col] = f2bf(hv > 0.f ? hv : 0.f);
            }
        }
        async_wait_all();
        __syncthreads();                      // h + sW2[fc] ready; sW1 reads done

        // prefetch next W1 chunk while phase 2 computes
        if (fc + 128 < F_DIM) {
            const unsigned short* src = w1e + (size_t)(fc + 128) * C_DIM;
            for (int u = threadIdx.x; u < 4096; u += 256) {
                int row = u >> 5, seg = u & 31;
                cp16_g2l(sW1 + row * W1_STRIDE + seg * 8, src + row * C_DIM + seg * 8);
            }
        }

        // ---- phase 2: Y[64 x 256] += h @ W2[fc:fc+128, :], A and B from LDS
        #pragma unroll
        for (int kk = 0; kk < 128; kk += 32) {
            v16bf a = load_a16(hsm + (wr * 16 + nloc) * H_STRIDE + kk, khalf);
            #pragma unroll
            for (int t = 0; t < 8; ++t) {
                int c = wc * 128 + t * 16 + nloc;
                v16bf b = *(const v16bf*)(sW2 + c * W2_STRIDE + kk + khalf * 16);
                acc2[t] = __builtin_amdgcn_wmma_f32_16x16x32_bf16(
                    false, a, false, b, (short)0, acc2[t], false, false);
            }
        }
        __syncthreads();                      // hsm/sW2 reads done before overwrite
    }
    // epilogue: + b2, scale by gate weight, scatter to ybuf[slot]
    #pragma unroll
    for (int t = 0; t < 8; ++t) {
        int c = wc * 128 + t * 16 + nloc;
        float bias = b2[e * C_DIM + c];
        #pragma unroll
        for (int v = 0; v < 8; ++v) {
            int row = wr * 16 + v + 8 * khalf;
            int slot = sslot[row];
            if (slot >= 0)
                ybuf[(size_t)slot * C_DIM + c] = swt[row] * (acc2[t][v] + bias);
        }
    }
}

__global__ void k_combine(const float* __restrict__ ybuf, float* __restrict__ outp, int n4) {
    int i = blockIdx.x * blockDim.x + threadIdx.x;
    if (i >= n4) return;
    size_t elem = (size_t)i * 4;
    size_t n = elem >> 8;            // / C
    size_t c = elem & (C_DIM - 1);
    float4 a = *(const float4*)(ybuf + (n * 2) * C_DIM + c);
    float4 b = *(const float4*)(ybuf + (n * 2 + 1) * C_DIM + c);
    float4 o; o.x = a.x + b.x; o.y = a.y + b.y; o.z = a.z + b.z; o.w = a.w + b.w;
    *(float4*)(outp + elem) = o;
}

// ---------------- host launch ----------------
extern "C" void kernel_launch(void* const* d_in, const int* in_sizes, int n_in,
                              void* d_out, int out_size, void* d_ws, size_t ws_size,
                              hipStream_t stream) {
    const float* x   = (const float*)d_in[0];
    const float* gW  = (const float*)d_in[1];
    const float* gb  = (const float*)d_in[2];
    const float* W1  = (const float*)d_in[3];
    const float* b1  = (const float*)d_in[4];
    const float* W2  = (const float*)d_in[5];
    const float* b2  = (const float*)d_in[6];
    float* outp = (float*)d_out;

    char* w = (char*)d_ws;
    size_t off = 0;
    auto take = [&](size_t bytes) { char* p = w + off; off = (off + bytes + 255) & ~(size_t)255; return p; };
    unsigned short* xb   = (unsigned short*)take((size_t)N_TOK * C_DIM * 2);          // 16 MB
    unsigned short* w1t  = (unsigned short*)take((size_t)E_NUM * F_DIM * C_DIM * 2);  // 8 MB
    unsigned short* w2t  = (unsigned short*)take((size_t)E_NUM * C_DIM * F_DIM * 2);  // 8 MB
    float* ybuf          = (float*)take((size_t)N_TOK * TOPK * C_DIM * 4);            // 64 MB
    int*   eidx          = (int*)take((size_t)N_TOK * TOPK * 4);
    int*   perm          = (int*)take((size_t)N_TOK * TOPK * 4);
    float* tw            = (float*)take((size_t)N_TOK * TOPK * 4);
    int*   counts        = (int*)take(64);
    int*   counts2       = (int*)take(64);
    int*   offA          = (int*)take(128);
    int*   poffA         = (int*)take(128);
    float* entpart       = (float*)take(GATE_BLOCKS * 4);
    (void)ws_size; (void)n_in; (void)in_sizes; (void)out_size;

    k_init<<<1, 32, 0, stream>>>(counts);
    k_cvt_x  <<<(N_TOK * C_DIM) / 256, 256, 0, stream>>>(x, xb, N_TOK * C_DIM);
    k_cvt_w1t<<<(E_NUM * C_DIM * F_DIM) / 256, 256, 0, stream>>>(W1, w1t);
    k_cvt_w2t<<<(E_NUM * F_DIM * C_DIM) / 256, 256, 0, stream>>>(W2, w2t);
    k_gate   <<<GATE_BLOCKS, 256, 0, stream>>>(x, gW, gb, eidx, tw, counts, entpart);
    k_scan   <<<1, 32, 0, stream>>>(counts, offA, poffA, counts2, entpart, outp);
    k_assign <<<(N_TOK * TOPK) / 256, 256, 0, stream>>>(eidx, offA, counts2, perm);
    k_moe    <<<MOE_TILES, 256, SMEM_BYTES, stream>>>(xb, w1t, w2t, b1, b2, offA, poffA,
                                                      counts, perm, tw, ybuf);
    k_combine<<<(N_TOK * C_DIM / 4 + 255) / 256, 256, 0, stream>>>(
        ybuf, outp, N_TOK * C_DIM / 4);
}
